// MambaBlock_49022756717214
// MI455X (gfx1250) — matrix operand order repaired and checked by reading, not compiled, not run on hardware
//
#include <hip/hip_runtime.h>
#include <hip/hip_bf16.h>

// ---------------------------------------------------------------------------
// CDNA5 (gfx1250) Mamba vision block.
// GEMMs via v_wmma_f32_16x16x32_bf16 (bf16 operands, f32 accumulate).
// A-tiles staged by the Tensor Data Mover (tensor_load_to_lds + TENSORcnt).
// fp32->bf16 via native __bf16 vector converts (packed cvt on CDNA5).
// ---------------------------------------------------------------------------

typedef __attribute__((ext_vector_type(16))) __bf16 v16bf;
typedef __attribute__((ext_vector_type(8)))  __bf16 v8bf;
typedef __attribute__((ext_vector_type(4)))  __bf16 v4bf;
typedef __attribute__((ext_vector_type(8)))  float  v8f;
typedef __attribute__((ext_vector_type(4)))  float  v4fv;
typedef __attribute__((ext_vector_type(4)))  unsigned int tdm_v4u;
typedef __attribute__((ext_vector_type(8)))  int          tdm_v8i;
typedef __attribute__((ext_vector_type(4)))  int          tdm_v4i;

// ---------------------------------------------------------------------------
// TDM: DMA a 2D f32 tile (tile_k x tile_m rows, row stride `stride` elems)
// from global memory into LDS at byte offset lds_off. LDS rows are padded
// 32 -> 36 DWORDs via D# pad (pad_interval code 4 = 32 DWORDs, pad_amount
// code 3 = 4 DWORDs) so fragment reads are bank-spread and 16B aligned.
// Issued by one wave; completion via s_wait_tensorcnt then workgroup barrier.
// D# layout per CDNA5 ISA.
// ---------------------------------------------------------------------------
__device__ __forceinline__ void tdm_load_tile_f32(const float* gsrc, unsigned lds_off,
                                                  int tile_k, int tile_m, int stride)
{
    const unsigned long long ga = (unsigned long long)(uintptr_t)gsrc;
    tdm_v4u g0;
    g0.x = 1u;                                              // count=1, user descriptor
    g0.y = lds_off;                                         // lds_addr (bytes)
    g0.z = (unsigned)(ga & 0xFFFFFFFFu);                    // global_addr[31:0]
    g0.w = (unsigned)((ga >> 32) & 0x01FFFFFFu) | (2u << 30); // global_addr[56:32] | type=2

    tdm_v8i g1;
    g1[0] = (int)((2u << 16) | (1u << 20) | (4u << 22) | (3u << 25)); // data_size=4B, pad en/int/amt
    g1[1] = (int)((unsigned)(tile_k & 0xFFFF) << 16);       // tensor_dim0[15:0] @ bits[63:48]
    g1[2] = (int)(((unsigned)tile_k >> 16) |                // tensor_dim0[31:16]
                  ((unsigned)(tile_m & 0xFFFF) << 16));     // tensor_dim1[15:0]
    g1[3] = (int)(((unsigned)tile_m >> 16) |                // tensor_dim1[31:16]
                  ((unsigned)(tile_k & 0xFFFF) << 16));     // tile_dim0 @ bits[127:112]
    g1[4] = (int)((unsigned)tile_m & 0xFFFF);               // tile_dim1 (tile_dim2=0)
    g1[5] = (int)(unsigned)stride;                          // tensor_dim0_stride[31:0]
    g1[6] = 0;                                              // stride[47:32] | dim1_stride lo
    g1[7] = 0;

    const tdm_v4i z4 = {0, 0, 0, 0};
#if defined(__clang_major__) && (__clang_major__ >= 23)
    const tdm_v8i z8 = {0, 0, 0, 0, 0, 0, 0, 0};
    __builtin_amdgcn_tensor_load_to_lds(g0, g1, z4, z4, z8, 0);
#else
    __builtin_amdgcn_tensor_load_to_lds(g0, g1, z4, z4, 0);
#endif
}

// ---------------------------------------------------------------------------
// WMMA GEMM:  C[M,N] = A[M,K] * Bw[N,K]^T
//   ATRANS:    A element (m,k) at A[((m/Lb)*KA + k)*Lb + (m%Lb)]  (in_proj);
//              staged via VGPRs (coalesced along m).
//   !ATRANS:   A element (m,k) at A[m*lda + k]; staged via TDM.
//   OUT_TRANS: C element (m,n) at C[((m/Lb)*ldc + n)*Lb + (m%Lb)] (out_proj NCHW)
// Weights valid for n<Nw, k<Kw; zero-padded in LDS beyond (handles K=48, N=56).
// Workgroup: 128 threads = 4 waves. Tile 64(M) x 64(N), K-step 32.
// Each wave owns a 16x64 strip -> 4 wmma accumulators.
// M % 64 == 0, N % 64 == 0, K % 32 == 0 (padded loop dims).
// ---------------------------------------------------------------------------
template <bool ATRANS, bool OUT_TRANS>
__global__ __launch_bounds__(128)
void gemm_wmma(const float* __restrict__ A, const float* __restrict__ Bw,
               float* __restrict__ C,
               int M, int N, int K, int Nw, int Kw,
               int ldbw, int lda, int ldc, int Lb, int KA)
{
    __shared__ __align__(16) float  Asf[64][36];  // [m][k] f32, stride 36 (TDM pad)
    __shared__ __align__(32) __bf16 Bs[32][80];   // [k][n] bf16, stride 80

    const int tid  = threadIdx.x;
    const int lane = tid & 31;
    const int wave = tid >> 5;            // 0..3
    const int m0   = blockIdx.y * 64;
    const int n0   = blockIdx.x * 64;

    v8f acc[4] = {};                       // 4 x (16x16 f32) accumulators

    for (int k0 = 0; k0 < K; k0 += 32) {
        // ---- stage A tile (64 x 32 f32) into LDS ---------------------------
        if (ATRANS) {
            const int b  = m0 / Lb;
            const int l0 = m0 % Lb;
            #pragma unroll
            for (int s = tid; s < 512; s += 128) {           // 32 k-rows x 16 l4-chunks
                const int kk = s >> 4;                        // 0..31
                const int l4 = (s & 15) << 2;                 // 0..60
                const v4fv v = *(const v4fv*)(A + ((size_t)b * KA + (k0 + kk)) * Lb + (l0 + l4));
                Asf[l4 + 0][kk] = v.x;
                Asf[l4 + 1][kk] = v.y;
                Asf[l4 + 2][kk] = v.z;
                Asf[l4 + 3][kk] = v.w;
            }
        } else {
            if (wave == 0) {
                tdm_load_tile_f32(A + (size_t)m0 * lda + k0,
                                  (unsigned)(uintptr_t)&Asf[0][0],
                                  /*tile_k=*/32, /*tile_m=*/64, /*stride=*/lda);
                __builtin_amdgcn_s_wait_tensorcnt(0);
            }
        }
        // ---- stage B tile (32 x 64) into LDS as bf16 (zero-pad beyond) -----
        #pragma unroll
        for (int s = tid; s < 512; s += 128) {               // 64 n-cols x 8 float4-chunks
            const int nn = s >> 3;                            // 0..63
            const int c4 = (s & 7) << 2;                      // 0..28
            const int gn = n0 + nn, gk = k0 + c4;
            v4fv v = {0.f, 0.f, 0.f, 0.f};
            if (gn < Nw && gk < Kw)
                v = *(const v4fv*)(Bw + (size_t)gn * ldbw + gk);
            const v4bf bv = __builtin_convertvector(v, v4bf); // packed f32->bf16 cvt
            Bs[c4 + 0][nn] = bv.x;
            Bs[c4 + 1][nn] = bv.y;
            Bs[c4 + 2][nn] = bv.z;
            Bs[c4 + 3][nn] = bv.w;
        }
        // prefetch next B tile stripe (emits global_prefetch_b8)
        if (k0 + 32 < Kw && (n0 + (tid & 63)) < Nw)
            __builtin_prefetch((const void*)(Bw + (size_t)(n0 + (tid & 63)) * ldbw + k0 + 32), 0, 0);

        __syncthreads();

        // ---- A fragment: lane&15 = M row; lane>>4 selects K half -----------
        // Elements 0..7  -> K = kh..kh+7 ; 8..15 -> K = 16+kh..16+kh+7
        const int mrow = 16 * wave + (lane & 15);
        const int kh   = (lane >> 4) << 3;                    // 0 or 8
        const v4bf b0 = __builtin_convertvector(*(const v4fv*)(&Asf[mrow][kh]),      v4bf);
        const v4bf b1 = __builtin_convertvector(*(const v4fv*)(&Asf[mrow][kh + 4]),  v4bf);
        const v4bf b2 = __builtin_convertvector(*(const v4fv*)(&Asf[mrow][kh + 16]), v4bf);
        const v4bf b3 = __builtin_convertvector(*(const v4fv*)(&Asf[mrow][kh + 20]), v4bf);
        const v8bf alo = __builtin_shufflevector(b0, b1, 0, 1, 2, 3, 4, 5, 6, 7);
        const v8bf ahi = __builtin_shufflevector(b2, b3, 0, 1, 2, 3, 4, 5, 6, 7);
        const v16bf afr = __builtin_shufflevector(alo, ahi,
            0, 1, 2, 3, 4, 5, 6, 7, 8, 9, 10, 11, 12, 13, 14, 15);

        // ---- B fragments: lane&15 = N col; K order same as A fragment ------
        #pragma unroll
        for (int nb = 0; nb < 4; ++nb) {
            v16bf bfr;
            #pragma unroll
            for (int s = 0; s < 16; ++s)
                bfr[s] = Bs[(s < 8) ? (kh + s) : (16 + kh + (s - 8))][16 * nb + (lane & 15)];
            acc[nb] = __builtin_amdgcn_wmma_f32_16x16x32_bf16(
                /*neg_a=*/false, afr, /*neg_b=*/false, bfr,
                /*c_mod=*/(short)0, acc[nb], /*reuse_a=*/false, /*reuse_b=*/false);
        }
        __syncthreads();
    }

    // ---- store: lane&15 = N col; VGPR r -> M row r + 8*(lane>>4) -----------
    const int nc    = lane & 15;
    const int mbase = m0 + 16 * wave + ((lane >> 4) << 3);
    #pragma unroll
    for (int nb = 0; nb < 4; ++nb) {
        const int gn = n0 + 16 * nb + nc;
        #pragma unroll
        for (int r = 0; r < 8; ++r) {
            const int gm = mbase + r;
            if (OUT_TRANS) {
                const int b = gm / Lb, l = gm % Lb;
                C[((size_t)b * ldc + gn) * Lb + l] = acc[nb][r];
            } else {
                C[(size_t)gm * ldc + gn] = acc[nb][r];
            }
        }
    }
}

// ---------------------------------------------------------------------------
// Depthwise 3x3 conv + bias + SiLU, channel-last (B, L=H*W, DI) layout.
// ---------------------------------------------------------------------------
__global__ __launch_bounds__(256)
void conv_silu_kernel(const float* __restrict__ x, const float* __restrict__ w,
                      const float* __restrict__ bias, float* __restrict__ out,
                      int Bn, int Hh, int Ww, int Ci)
{
    const int idx = blockIdx.x * blockDim.x + threadIdx.x;
    const int HW  = Hh * Ww;
    if (idx >= Bn * HW * Ci) return;
    const int di = idx % Ci;
    const int l  = (idx / Ci) % HW;
    const int b  = idx / (Ci * HW);
    const int h  = l / Ww, wq = l % Ww;

    float acc = bias[di];
    #pragma unroll
    for (int dy = 0; dy < 3; ++dy) {
        const int hh = h + dy - 1;
        if (hh < 0 || hh >= Hh) continue;
        #pragma unroll
        for (int dx = 0; dx < 3; ++dx) {
            const int ww = wq + dx - 1;
            if (ww < 0 || ww >= Ww) continue;
            acc += x[((size_t)b * HW + (size_t)hh * Ww + ww) * Ci + di] * w[di * 9 + dy * 3 + dx];
        }
    }
    out[idx] = acc / (1.f + expf(-acc));   // SiLU
}

// ---------------------------------------------------------------------------
// Selective scan: one thread per (b, di); 4 states; fuses dt-bias + softplus.
//   delta  : (B, L, DI) raw GEMM result
//   xsc    : (B, L, DI) post conv+SiLU (u)
//   xdbl   : (B, L, 64) padded x_proj output; Bs = cols 48..51, Cs = 52..55
//   y      : (B, L, DI)
// ---------------------------------------------------------------------------
__global__ __launch_bounds__(256)
void scan_kernel(const float* __restrict__ delta, const float* __restrict__ xsc,
                 const float* __restrict__ xdbl, const float* __restrict__ dt_b,
                 const float* __restrict__ A_log, const float* __restrict__ D_param,
                 float* __restrict__ y, int Bn, int Lq, int Di)
{
    const int idx = blockIdx.x * blockDim.x + threadIdx.x;
    if (idx >= Bn * Di) return;
    const int di = idx % Di;
    const int b  = idx / Di;

    float An[4];
    #pragma unroll
    for (int n = 0; n < 4; ++n) An[n] = -expf(A_log[di * 4 + n]);
    const float Dp  = D_param[di];
    const float dtb = dt_b[di];

    const float* dp = delta + (size_t)b * Lq * Di + di;
    const float* up = xsc   + (size_t)b * Lq * Di + di;
    const float* xd = xdbl  + (size_t)b * Lq * 64;
    float*       yp = y     + (size_t)b * Lq * Di + di;

    float s[4] = {0.f, 0.f, 0.f, 0.f};
    for (int l = 0; l < Lq; ++l) {
        float d = dp[(size_t)l * Di] + dtb;
        d = (d > 20.f) ? d : log1pf(expf(d));          // softplus
        const float u  = up[(size_t)l * Di];
        const float du = d * u;
        const float* row = xd + (size_t)l * 64;
        float acc = 0.f;
        #pragma unroll
        for (int n = 0; n < 4; ++n) {
            s[n] = expf(d * An[n]) * s[n] + du * row[48 + n];
            acc += s[n] * row[52 + n];
        }
        yp[(size_t)l * Di] = acc + Dp * u;
    }
}

// ---------------------------------------------------------------------------
// Row LayerNorm over DI: one 256-thread block per row; wave32 shfl + LDS reduce.
// ---------------------------------------------------------------------------
__global__ __launch_bounds__(256)
void layernorm_kernel(const float* __restrict__ y, const float* __restrict__ g,
                      const float* __restrict__ bta, float* __restrict__ out, int Di)
{
    const int row = blockIdx.x;
    const float* yr = y + (size_t)row * Di;
    float sum = 0.f, sq = 0.f;
    for (int i = threadIdx.x; i < Di; i += 256) { const float v = yr[i]; sum += v; sq += v * v; }

    #pragma unroll
    for (int o = 16; o > 0; o >>= 1) {
        sum += __shfl_down(sum, o, 32);
        sq  += __shfl_down(sq,  o, 32);
    }
    __shared__ float ssum[8], ssq[8];
    __shared__ float smean, srstd;
    const int lane = threadIdx.x & 31, wid = threadIdx.x >> 5;
    if (lane == 0) { ssum[wid] = sum; ssq[wid] = sq; }
    __syncthreads();
    if (threadIdx.x == 0) {
        float ts = 0.f, tq = 0.f;
        #pragma unroll
        for (int i = 0; i < 8; ++i) { ts += ssum[i]; tq += ssq[i]; }
        const float mean = ts / (float)Di;
        smean = mean;
        srstd = rsqrtf(tq / (float)Di - mean * mean + 1e-5f);
    }
    __syncthreads();
    const float mean = smean, rstd = srstd;
    for (int i = threadIdx.x; i < Di; i += 256)
        out[(size_t)row * Di + i] = (yr[i] - mean) * rstd * g[i] + bta[i];
}

// ---------------------------------------------------------------------------
// Launch: hs->in_proj->conv/silu->x_proj->delta->scan->LN->out_proj
// ---------------------------------------------------------------------------
extern "C" void kernel_launch(void* const* d_in, const int* in_sizes, int n_in,
                              void* d_out, int out_size, void* d_ws, size_t ws_size,
                              hipStream_t stream)
{
    const float* hs   = (const float*)d_in[0];   // (B, DM, H, W)
    const float* wip  = (const float*)d_in[1];   // (DI, DM)
    const float* cw   = (const float*)d_in[2];   // (DI, 1, 3, 3)
    const float* cb   = (const float*)d_in[3];   // (DI)
    const float* wxp  = (const float*)d_in[4];   // (R+2N=56, DI)
    const float* dtw  = (const float*)d_in[5];   // (1, DI, R)
    const float* dtb  = (const float*)d_in[6];   // (1, DI)
    const float* alog = (const float*)d_in[7];   // (1, DI, 4)
    const float* dpar = (const float*)d_in[8];   // (1, DI)
    const float* lng  = (const float*)d_in[9];   // (DI)
    const float* lnb  = (const float*)d_in[10];  // (DI)
    const float* wop  = (const float*)d_in[11];  // (DM, DI)
    float* out = (float*)d_out;                  // (B, DM, H, W)

    const int Bn = 2, DM = 768, Hh = 64, Ww = 64, Lq = 4096, Di = 1536, Rr = 48;
    const int Mrows = Bn * Lq;                   // 8192

    const size_t SZB = (size_t)Bn * Lq * Di * sizeof(float);  // 50.3 MB
    char* ws = (char*)d_ws;
    float* bufX  = (float*)(ws);                 // in_proj out, later raw delta
    float* bufXC = (float*)(ws + SZB);           // conv+SiLU out, later y_ln
    float* bufY  = (float*)(ws + 2 * SZB);       // scan out
    float* bufXD = (float*)(ws + 3 * SZB);       // x_dbl (8192 x 64, padded)

    const dim3 blk(128);

    // 1. in_proj: x[b,l,di] = sum_dm hs[b,dm,l] * Wip[di,dm]   (A transposed)
    gemm_wmma<true, false><<<dim3(Di / 64, Mrows / 64), blk, 0, stream>>>(
        hs, wip, bufX, Mrows, Di, /*K=*/DM, /*Nw=*/Di, /*Kw=*/DM,
        /*ldbw=*/DM, /*lda=*/0, /*ldc=*/Di, /*Lb=*/Lq, /*KA=*/DM);

    // 2. depthwise 3x3 conv + bias + SiLU (channel-last)
    {
        const int total = Bn * Lq * Di;
        conv_silu_kernel<<<(total + 255) / 256, 256, 0, stream>>>(
            bufX, cw, cb, bufXC, Bn, Hh, Ww, Di);
    }

    // 3. x_proj: x_dbl[m, 0..55] = xsc[m,:] * Wxp^T   (N padded 56 -> 64)
    gemm_wmma<false, false><<<dim3(1, Mrows / 64), blk, 0, stream>>>(
        bufXC, wxp, bufXD, Mrows, /*N=*/64, /*K=*/Di, /*Nw=*/56, /*Kw=*/Di,
        /*ldbw=*/Di, /*lda=*/Di, /*ldc=*/64, /*Lb=*/Lq, /*KA=*/0);

    // 4. delta: delta[m,di] = sum_r x_dbl[m,r] * dt_w[di,r]   (K padded 48 -> 64)
    gemm_wmma<false, false><<<dim3(Di / 64, Mrows / 64), blk, 0, stream>>>(
        bufXD, dtw, bufX, Mrows, /*N=*/Di, /*K=*/64, /*Nw=*/Di, /*Kw=*/Rr,
        /*ldbw=*/Rr, /*lda=*/64, /*ldc=*/Di, /*Lb=*/Lq, /*KA=*/0);

    // 5. selective scan (fused dt-bias + softplus + recurrence + D*u)
    scan_kernel<<<(Bn * Di + 255) / 256, 256, 0, stream>>>(
        bufX, bufXC, bufXD, dtb, alog, dpar, bufY, Bn, Lq, Di);

    // 6. LayerNorm over DI (y -> y_ln in bufXC, which is free after the scan)
    layernorm_kernel<<<Mrows, 256, 0, stream>>>(bufY, lng, lnb, bufXC, Di);

    // 7. out_proj with NCHW scatter: out[b,dm,l] = sum_di y_ln[b,l,di]*Wop[dm,di]
    gemm_wmma<false, true><<<dim3(DM / 64, Mrows / 64), blk, 0, stream>>>(
        bufXC, wop, out, Mrows, /*N=*/DM, /*K=*/Di, /*Nw=*/DM, /*Kw=*/Di,
        /*ldbw=*/Di, /*lda=*/Di, /*ldc=*/DM, /*Lb=*/Lq, /*KA=*/0);
}
